// Model_66211215835668
// MI455X (gfx1250) — matrix operand order repaired and checked by reading, not compiled, not run on hardware
//
#include <hip/hip_runtime.h>
#include <hip/hip_bf16.h>
#include <math.h>

// MI455X / gfx1250: wave32, WMMA f32_16x16x32_f16 for all GEMMs, f32 accumulate.

typedef __attribute__((ext_vector_type(16))) _Float16 v16h;
typedef __attribute__((ext_vector_type(8)))  _Float16 v8h;
typedef __attribute__((ext_vector_type(4)))  _Float16 v4h;
typedef __attribute__((ext_vector_type(8)))  float    v8f;

#define BS   4096
#define FD   768
#define PD   500
#define ZD   300
#define RNUM 10
#define UNUM 10
#define NPER 33   // 3 + 2*R + U
#define KC   512  // K-chunk staged in LDS (32 rows x 512 halves = 32 KB)

// ---------------------------------------------------------------------------
// Pack W [N,K] f32 into f16 WMMA-B layout, zero-padded in both N (to NtilesPad
// tiles, covering the GEMM grid exactly so the K-loop needs no guards) and K:
//   Wp[((tn*nks + ks)*32 + lane)*16 + e] = W[tn*16 + lane%16, ks*32 + (lane/16)*16 + e]
// A lane's 16 B-halves for (tile,k-step) are 32 contiguous bytes.
// ---------------------------------------------------------------------------
__global__ void pack_w_f16(const float* __restrict__ W, _Float16* __restrict__ Wp,
                           int N, int K, int NtilesPad, int nks)
{
    const long g = (long)blockIdx.x * blockDim.x + threadIdx.x;
    const long total = (long)NtilesPad * nks * 32;
    if (g >= total) return;
    const int lane = (int)(g & 31);
    const long tk  = g >> 5;                 // tn*nks + ks
    const int ks = (int)(tk % nks);
    const int tn = (int)(tk / nks);
    const int n  = tn * 16 + (lane & 15);
    const int k0 = ks * 32 + (lane >> 4) * 16;
    _Float16* dst = Wp + g * 16;
#pragma unroll
    for (int e = 0; e < 16; ++e) {
        const int k = k0 + e;
        const float v = (n < N && k < K) ? W[(long)n * K + k] : 0.f;
        dst[e] = (_Float16)v;
    }
}

// ---------------------------------------------------------------------------
// Fused GEMM: out[map(i), n] = act( sum_k A[i,k] * W[n,k] + bias[n] )
//   A: [M,K] f32 row-major (M % 32 == 0, K % 4 == 0), Wp: packed f16 B
//   (zero-padded to the full grid, so the hot loop is branch-free).
//   map(i) = (i / rps) * row_mult + off + (i % rps); act: 0=none,1=tanh,2=relu.
// Block = 128 threads (4 waves) -> 32(M) x (4*TPW*16)(N) output tile.
//   A staged in LDS as f16 in KC-chunks (read from HBM exactly once).
//   Per 32-K step: 4 ds_load_b128 (A frags) + 2*TPW global_load_b128 (B, from
//   L2-resident packed W, all clause-able) + 2*TPW back-to-back v_wmma.
// ---------------------------------------------------------------------------
template<int TPW>
__global__ __launch_bounds__(128)
void wmma_gemm_act(const float* __restrict__ A, const _Float16* __restrict__ Wp,
                   const float* __restrict__ bias, float* __restrict__ out,
                   int N, int K, int Kpad, int nks, int ldout,
                   int rps, int row_mult, int off, int act)
{
    __shared__ _Float16 sA[32 * KC];
    const int tid  = threadIdx.x;
    const int lane = tid & 31;
    const int wave = tid >> 5;
    const int h16  = lane >> 4;
    const int l16  = lane & 15;
    const long row0 = (long)blockIdx.x * 32;

    // wave-uniform first N-tile index (scalar)
    const int tn0 = __builtin_amdgcn_readfirstlane(((int)blockIdx.y * 4 + wave) * TPW);

    const v8f vz = {0.f,0.f,0.f,0.f,0.f,0.f,0.f,0.f};
    v8f acc0[TPW], acc1[TPW];
#pragma unroll
    for (int t = 0; t < TPW; ++t) { acc0[t] = vz; acc1[t] = vz; }

    // per-lane base into packed W for this wave's first tile
    const _Float16* wbase = Wp + (long)tn0 * nks * 512 + lane * 16;

    for (int kc = 0; kc < Kpad; kc += KC) {
        const int kcw = (Kpad - kc) < KC ? (Kpad - kc) : KC;   // multiple of 32

        // ---- stage 32 x kcw A sub-slab: f32 global -> f16 LDS, zero-padded ----
        const int kq = kcw >> 2;
        for (int idx = tid; idx < 32 * kq; idx += 128) {
            const int r  = idx / kq;
            const int kl = (idx - r * kq) << 2;
            const int k  = kc + kl;
            v4h h;
            if (k < K) {                     // K%4==0 -> float4 fully in-bounds
                const float4 f = *(const float4*)(A + (row0 + r) * (long)K + k);
                h.x = (_Float16)f.x; h.y = (_Float16)f.y;
                h.z = (_Float16)f.z; h.w = (_Float16)f.w;
            } else {
                h.x = (_Float16)0.f; h.y = (_Float16)0.f;
                h.z = (_Float16)0.f; h.w = (_Float16)0.f;
            }
            *(v4h*)(sA + r * KC + kl) = h;
        }
        __syncthreads();

        for (int kl = 0; kl < kcw; kl += 32) {
            const int ksg = (kc + kl) >> 5;  // global k-step index
            // A fragments (contiguous layout: element e -> K = h16*16 + e, matching packed B)
            const v8h a0lo = *(const v8h*)(sA + l16 * KC + kl + h16 * 16);
            const v8h a0hi = *(const v8h*)(sA + l16 * KC + kl + 8 + h16 * 16);
            const v8h a1lo = *(const v8h*)(sA + (16 + l16) * KC + kl + h16 * 16);
            const v8h a1hi = *(const v8h*)(sA + (16 + l16) * KC + kl + 8 + h16 * 16);
            const v16h a0 = __builtin_shufflevector(a0lo, a0hi,
                0,1,2,3,4,5,6,7,8,9,10,11,12,13,14,15);
            const v16h a1 = __builtin_shufflevector(a1lo, a1hi,
                0,1,2,3,4,5,6,7,8,9,10,11,12,13,14,15);

            // gather all B fragments first (one big clause, max MLP), then WMMA burst
            v16h bfr[TPW];
#pragma unroll
            for (int t = 0; t < TPW; ++t) {
                const _Float16* p = wbase + ((long)t * nks + ksg) * 512;
                const v8h b0 = *(const v8h*)p;
                const v8h b1 = *(const v8h*)(p + 8);
                bfr[t] = __builtin_shufflevector(b0, b1,
                    0,1,2,3,4,5,6,7,8,9,10,11,12,13,14,15);
            }
#pragma unroll
            for (int t = 0; t < TPW; ++t) {
                acc0[t] = __builtin_amdgcn_wmma_f32_16x16x32_f16(
                    false, a0, false, bfr[t], (short)0, acc0[t], false, false);
                acc1[t] = __builtin_amdgcn_wmma_f32_16x16x32_f16(
                    false, a1, false, bfr[t], (short)0, acc1[t], false, false);
            }
        }
        __syncthreads();
    }

    // ---- epilogue: bias + activation + scattered f32 stores (both M-tiles) ----
    long orow0[8], orow1[8];
#pragma unroll
    for (int r = 0; r < 8; ++r) {
        const long i0 = row0 + r + 8 * h16;        // C/D layout: M = r + 8*h16
        const long i1 = row0 + 16 + r + 8 * h16;
        orow0[r] = (i0 / rps) * (long)row_mult + off + (i0 % rps);
        orow1[r] = (i1 / rps) * (long)row_mult + off + (i1 % rps);
    }
#pragma unroll
    for (int t = 0; t < TPW; ++t) {
        const int n = (tn0 + t) * 16 + l16;
        if (n >= N) continue;                      // padded tiles map to n >= N
        const float bv = bias[n];
#pragma unroll
        for (int r = 0; r < 8; ++r) {
            float v0 = acc0[t][r] + bv;
            float v1 = acc1[t][r] + bv;
            if (act == 1)      { v0 = tanhf(v0);      v1 = tanhf(v1); }
            else if (act == 2) { v0 = fmaxf(v0, 0.f); v1 = fmaxf(v1, 0.f); }
            out[orow0[r] * (long)ldout + n] = v0;
            out[orow1[r] * (long)ldout + n] = v1;
        }
    }
}

// ---------------------------------------------------------------------------
// Hypergraph conv (after theta): edge means -> node means -> relu.
// Per sample b: edges e0={0,1,2}, e1={3..12,0}, e2={13..22,1}, e3={23..32,2}.
// 10 extra pair edges {3+i,13+i} hit sample 0 only (bias bug in reference).
// ---------------------------------------------------------------------------
__global__ void hgconv_kernel(const float* __restrict__ Xt, float* __restrict__ Xo)
{
    const int b = blockIdx.x;
    const float* xt = Xt + (long)b * NPER * ZD;
    float* xo = Xo + (long)b * NPER * ZD;
    for (int d = threadIdx.x; d < ZD; d += blockDim.x) {
        const float x0 = xt[0*ZD+d], x1 = xt[1*ZD+d], x2 = xt[2*ZD+d];
        float srt = 0.f, srv = 0.f, sru = 0.f;
#pragma unroll
        for (int r = 0; r < RNUM; ++r) {
            srt += xt[(3 + r) * ZD + d];
            srv += xt[(3 + RNUM + r) * ZD + d];
        }
#pragma unroll
        for (int u = 0; u < UNUM; ++u) sru += xt[(3 + 2*RNUM + u) * ZD + d];
        const float Y0 = (x0 + x1 + x2) * (1.f / 3.f);
        const float Y1 = (srt + x0) * (1.f / 11.f);
        const float Y2 = (srv + x1) * (1.f / 11.f);
        const float Y3 = (sru + x2) * (1.f / 11.f);
        xo[0*ZD+d] = fmaxf(0.5f * (Y0 + Y1), 0.f);
        xo[1*ZD+d] = fmaxf(0.5f * (Y0 + Y2), 0.f);
        xo[2*ZD+d] = fmaxf(0.5f * (Y0 + Y3), 0.f);
        if (b != 0) {
#pragma unroll
            for (int r = 0; r < RNUM; ++r) {
                xo[(3 + r) * ZD + d]        = fmaxf(Y1, 0.f);
                xo[(3 + RNUM + r) * ZD + d] = fmaxf(Y2, 0.f);
            }
        } else {
#pragma unroll
            for (int r = 0; r < RNUM; ++r) {
                const float xa = xt[(3 + r) * ZD + d];
                const float xb = xt[(3 + RNUM + r) * ZD + d];
                const float Yp = 0.5f * (xa + xb);
                xo[(3 + r) * ZD + d]        = fmaxf(0.5f * (Y1 + Yp), 0.f);
                xo[(3 + RNUM + r) * ZD + d] = fmaxf(0.5f * (Y2 + Yp), 0.f);
            }
        }
#pragma unroll
        for (int u = 0; u < UNUM; ++u)
            xo[(3 + 2*RNUM + u) * ZD + d] = fmaxf(Y3, 0.f);
    }
}

// ---------------------------------------------------------------------------
// Softmax-weighted aggregation + feature assembly.
// feat = [Xo0 | Xo1 | Xo2 | rv_agg | rt_agg | ru_agg | lbl_emb]  (2100 floats)
// ---------------------------------------------------------------------------
__global__ void agg_feat_kernel(const float* __restrict__ Xo,
                                const float* __restrict__ sim,
                                const float* __restrict__ usim,
                                const float* __restrict__ rlabel,
                                const float* __restrict__ lbl_w,
                                const float* __restrict__ lbl_b,
                                float* __restrict__ feat)
{
    const int b = blockIdx.x;
    float w[RNUM], uw[UNUM];
    float m1 = -1e30f, m2 = -1e30f;
#pragma unroll
    for (int r = 0; r < RNUM; ++r) { w[r]  = sim[b * RNUM + r];  m1 = fmaxf(m1, w[r]);  }
#pragma unroll
    for (int u = 0; u < UNUM; ++u) { uw[u] = usim[b * UNUM + u]; m2 = fmaxf(m2, uw[u]); }
    float s1 = 0.f, s2 = 0.f;
#pragma unroll
    for (int r = 0; r < RNUM; ++r) { w[r]  = expf(w[r]  - m1); s1 += w[r];  }
#pragma unroll
    for (int u = 0; u < UNUM; ++u) { uw[u] = expf(uw[u] - m2); s2 += uw[u]; }
    const float i1 = 1.f / s1, i2 = 1.f / s2;
    float lbl_agg = 0.f;
#pragma unroll
    for (int r = 0; r < RNUM; ++r) { w[r] *= i1; lbl_agg += w[r] * rlabel[b * RNUM + r]; }
#pragma unroll
    for (int u = 0; u < UNUM; ++u) uw[u] *= i2;

    const float* xo = Xo + (long)b * NPER * ZD;
    float* f = feat + (long)b * (7 * ZD);
    for (int d = threadIdx.x; d < ZD; d += blockDim.x) {
        f[0*ZD + d] = xo[0*ZD + d];
        f[1*ZD + d] = xo[1*ZD + d];
        f[2*ZD + d] = xo[2*ZD + d];
        float rt = 0.f, rv = 0.f, ru = 0.f;
#pragma unroll
        for (int r = 0; r < RNUM; ++r) {
            rt += w[r]  * xo[(3 + r) * ZD + d];
            rv += w[r]  * xo[(3 + RNUM + r) * ZD + d];
            ru += uw[r] * xo[(3 + 2*RNUM + r) * ZD + d];
        }
        f[3*ZD + d] = rv;                               // rv_agg first!
        f[4*ZD + d] = rt;
        f[5*ZD + d] = ru;
        f[6*ZD + d] = fmaxf(lbl_agg * lbl_w[d] + lbl_b[d], 0.f);
    }
}

// ---------------------------------------------------------------------------
// Final layer: out[b] = sigmoid(dot(h2[b,:200], p3_w) + p3_b). Wave per sample.
// ---------------------------------------------------------------------------
__global__ void final_kernel(const float* __restrict__ h2,
                             const float* __restrict__ p3_w,
                             const float* __restrict__ p3_b,
                             float* __restrict__ out, int bs)
{
    const int wid  = (int)((blockIdx.x * blockDim.x + threadIdx.x) >> 5);
    const int lane = threadIdx.x & 31;
    if (wid >= bs) return;
    const float* h = h2 + (long)wid * 200;
    float s = 0.f;
    for (int k = lane; k < 200; k += 32) s += h[k] * p3_w[k];
#pragma unroll
    for (int o = 16; o > 0; o >>= 1) s += __shfl_xor(s, o, 32);
    if (lane == 0) out[wid] = 1.f / (1.f + expf(-(s + p3_b[0])));
}

// ---------------------------------------------------------------------------
static inline int ntiles_of(int N) { return (N + 15) / 16; }
static inline int nks_of(int K)    { return ((K + 31) & ~31) / 32; }

// grid.y blocks of 4 waves x TPW tiles; packed W padded to full coverage
static inline int gridy_of(int N, int TPW) {
    return (ntiles_of(N) + 4 * TPW - 1) / (4 * TPW);
}
static inline int ntilespad_of(int N, int TPW) { return gridy_of(N, TPW) * 4 * TPW; }

template<int TPW>
static void gemm_launch(const float* A, const _Float16* Wp, const float* bias,
                        float* out, int M, int N, int K, int rps, int row_mult,
                        int off, int act, int ldout, hipStream_t stream)
{
    const int Kpad = (K + 31) & ~31;
    const int nks = nks_of(K);
    dim3 grid(M / 32, gridy_of(N, TPW));
    wmma_gemm_act<TPW><<<grid, 128, 0, stream>>>(
        A, Wp, bias, out, N, K, Kpad, nks, ldout, rps, row_mult, off, act);
}

extern "C" void kernel_launch(void* const* d_in, const int* in_sizes, int n_in,
                              void* d_out, int out_size, void* d_ws, size_t ws_size,
                              hipStream_t stream)
{
    (void)in_sizes; (void)n_in; (void)out_size; (void)ws_size;
    const float* visual  = (const float*)d_in[0];
    const float* textual = (const float*)d_in[1];
    const float* sim     = (const float*)d_in[2];
    const float* r_vis   = (const float*)d_in[3];   // [bs,10,1,768] == [bs*10,768]
    const float* r_txt   = (const float*)d_in[4];
    const float* r_lbl   = (const float*)d_in[5];
    const float* user    = (const float*)d_in[6];
    const float* r_usr   = (const float*)d_in[7];
    const float* r_usim  = (const float*)d_in[8];
    const float* vis_w  = (const float*)d_in[9];  const float* vis_b  = (const float*)d_in[10];
    const float* txt_w  = (const float*)d_in[11]; const float* txt_b  = (const float*)d_in[12];
    const float* usr_w  = (const float*)d_in[13]; const float* usr_b  = (const float*)d_in[14];
    const float* rvis_w = (const float*)d_in[15]; const float* rvis_b = (const float*)d_in[16];
    const float* rtxt_w = (const float*)d_in[17]; const float* rtxt_b = (const float*)d_in[18];
    const float* rusr_w = (const float*)d_in[19]; const float* rusr_b = (const float*)d_in[20];
    const float* hg_w   = (const float*)d_in[21]; const float* hg_b   = (const float*)d_in[22];
    const float* lbl_w  = (const float*)d_in[23]; const float* lbl_b  = (const float*)d_in[24];
    const float* p1_w   = (const float*)d_in[25]; const float* p1_b   = (const float*)d_in[26];
    const float* p2_w   = (const float*)d_in[27]; const float* p2_b   = (const float*)d_in[28];
    const float* p3_w   = (const float*)d_in[29]; const float* p3_b   = (const float*)d_in[30];

    // ---- workspace carve-out (~495 MB) ----
    char* ws = (char*)d_ws;
    size_t woff = 0;
    auto carve = [&](size_t bytes) -> void* {
        void* p = ws + woff;
        woff += (bytes + 255) & ~(size_t)255;
        return p;
    };
    float* X    = (float*)carve((size_t)BS * NPER * PD * sizeof(float)); // 270 MB
    float* Xt   = (float*)carve((size_t)BS * NPER * ZD * sizeof(float)); // 162 MB
    float* feat = (float*)carve((size_t)BS * 7 * ZD * sizeof(float));
    float* h1   = (float*)carve((size_t)BS * 800 * sizeof(float));
    float* h2   = (float*)carve((size_t)BS * 200 * sizeof(float));
    float* Xo   = X;  // X is dead once Xt is computed; reuse its space

    auto packed_sz = [&](int N, int K, int TPW) {
        return (size_t)ntilespad_of(N, TPW) * nks_of(K) * 32 * 16 * sizeof(_Float16);
    };
    // TPW per GEMM shape: embed 8 (32 tiles, 0 pad), theta 5 (20/19),
    // p1 7 (56/50, 2 blocks), p2 4 (16/13)
    _Float16* Wtxt = (_Float16*)carve(packed_sz(PD, FD, 8));
    _Float16* Wvis = (_Float16*)carve(packed_sz(PD, FD, 8));
    _Float16* Wusr = (_Float16*)carve(packed_sz(PD, FD, 8));
    _Float16* Wrt  = (_Float16*)carve(packed_sz(PD, FD, 8));
    _Float16* Wrv  = (_Float16*)carve(packed_sz(PD, FD, 8));
    _Float16* Wru  = (_Float16*)carve(packed_sz(PD, FD, 8));
    _Float16* Whg  = (_Float16*)carve(packed_sz(ZD, PD, 5));
    _Float16* Wp1  = (_Float16*)carve(packed_sz(800, 7 * ZD, 7));
    _Float16* Wp2  = (_Float16*)carve(packed_sz(200, 800, 4));

    auto pack = [&](const float* W, _Float16* Wp, int N, int K, int TPW) {
        const int ntp = ntilespad_of(N, TPW), nks = nks_of(K);
        const long groups = (long)ntp * nks * 32;
        pack_w_f16<<<(unsigned)((groups + 255) / 256), 256, 0, stream>>>(
            W, Wp, N, K, ntp, nks);
    };
    pack(txt_w,  Wtxt, PD, FD, 8);
    pack(vis_w,  Wvis, PD, FD, 8);
    pack(usr_w,  Wusr, PD, FD, 8);
    pack(rtxt_w, Wrt,  PD, FD, 8);
    pack(rvis_w, Wrv,  PD, FD, 8);
    pack(rusr_w, Wru,  PD, FD, 8);
    pack(hg_w,   Whg,  ZD, PD, 5);
    pack(p1_w,   Wp1,  800, 7 * ZD, 7);
    pack(p2_w,   Wp2,  200, 800, 4);

    // 6 tanh embedding GEMMs scattered into X[bs,33,500] (node order t,v,u,rt,rv,ru)
    gemm_launch<8>(textual, Wtxt, txt_b,  X, BS,      PD, FD, 1,    NPER, 0,          1, PD, stream);
    gemm_launch<8>(visual,  Wvis, vis_b,  X, BS,      PD, FD, 1,    NPER, 1,          1, PD, stream);
    gemm_launch<8>(user,    Wusr, usr_b,  X, BS,      PD, FD, 1,    NPER, 2,          1, PD, stream);
    gemm_launch<8>(r_txt,   Wrt,  rtxt_b, X, BS*RNUM, PD, FD, RNUM, NPER, 3,          1, PD, stream);
    gemm_launch<8>(r_vis,   Wrv,  rvis_b, X, BS*RNUM, PD, FD, RNUM, NPER, 3 + RNUM,   1, PD, stream);
    gemm_launch<8>(r_usr,   Wru,  rusr_b, X, BS*UNUM, PD, FD, UNUM, NPER, 3 + 2*RNUM, 1, PD, stream);

    // theta: Xt = X @ hg_w^T + hg_b   [135168, 500] -> [135168, 300]
    gemm_launch<5>(X, Whg, hg_b, Xt, BS * NPER, ZD, PD, 1, 1, 0, 0, ZD, stream);

    hgconv_kernel<<<BS, 128, 0, stream>>>(Xt, Xo);
    agg_feat_kernel<<<BS, 128, 0, stream>>>(Xo, sim, r_usim, r_lbl, lbl_w, lbl_b, feat);

    // MLP head
    gemm_launch<7>(feat, Wp1, p1_b, h1, BS, 800, 7 * ZD, 1, 1, 0, 2, 800, stream);
    gemm_launch<4>(h1,   Wp2, p2_b, h2, BS, 200, 800,    1, 1, 0, 2, 200, stream);
    final_kernel<<<BS / 8, 256, 0, stream>>>(h2, p3_w, p3_b, (float*)d_out, BS);
}
